// TGN_29850022707223
// MI455X (gfx1250) — matrix operand order repaired and checked by reading, not compiled, not run on hardware
//
#include <hip/hip_runtime.h>
#include <hip/hip_bf16.h>
#include <math.h>

#define NUM_NODES 200000
#define BATCH     65536
#define NEVT      131072          // E
#define NE2       (2 * NEVT)      // 262144
#define DMEM      128
#define DIN       512             // 2*D_MEM + D_RAW + D_TIME
#define DH        128
#define G3        384             // 3*D_MEM
#define MROWS     32              // M-tile per block (2 WMMA row-subtiles)

typedef __attribute__((ext_vector_type(16))) __bf16 v16bf;
typedef __attribute__((ext_vector_type(8)))  float  v8f;

union ABFrag { uint4 u[2]; v16bf v; };

__device__ __forceinline__ unsigned short f2bf(float x) {
  unsigned int u = __float_as_uint(x);
  u += 0x7fffu + ((u >> 16) & 1u);      // round-to-nearest-even
  return (unsigned short)(u >> 16);
}

// ---------------- kernel 1: assoc scatter + maxkey init ----------------
__global__ void tgn_init(const int* __restrict__ n_id,
                         int* __restrict__ assoc,
                         unsigned long long* __restrict__ maxkey) {
  int i = blockIdx.x * blockDim.x + threadIdx.x;
  if (i < BATCH) {
    assoc[n_id[i]] = i;
    maxkey[i] = 0ull;
  }
}

// ---------------- kernel 2: LastAggregator argmax over events ----------------
__global__ void tgn_scan(const int* __restrict__ src_s, const int* __restrict__ dst_d,
                         const int* __restrict__ t_s,  const int* __restrict__ t_d,
                         const int* __restrict__ assoc,
                         unsigned long long* __restrict__ maxkey) {
  int e = blockIdx.x * blockDim.x + threadIdx.x;
  if (e >= NE2) return;
  int g, t;
  if (e < NEVT) { g = src_s[e];        t = t_s[e]; }
  else          { g = dst_d[e - NEVT]; t = t_d[e - NEVT]; }
  int loc = assoc[g];
  unsigned long long key =
      (unsigned long long)(unsigned int)t * (unsigned long long)NE2 +
      (unsigned long long)e + 1ull;     // +1 so 0 == "no event"
  atomicMax(&maxkey[loc], key);
}

// ---------------- kernel 3: weights -> bf16 ----------------
__global__ void tgn_wcvt(const float* __restrict__ W_ih, const float* __restrict__ W_hh,
                         unsigned short* __restrict__ Wih_bf,
                         unsigned short* __restrict__ Whh_bf) {
  int i = blockIdx.x * blockDim.x + threadIdx.x;
  if (i < G3 * DIN) {
    Wih_bf[i] = f2bf(W_ih[i]);
  } else {
    int j = i - G3 * DIN;
    if (j < G3 * DH) Whh_bf[j] = f2bf(W_hh[j]);
  }
}

// ---------------- WMMA inner GEMM: 2 row-subtiles x 6 n-tiles, constant K ----------------
template <int K, int ABASE>
__device__ __forceinline__ void gemm_part(const unsigned short (*sA)[DIN + DH],
                                          const unsigned short* __restrict__ Wmat,
                                          int tbase, int m, int hi, v8f acc[2][6]) {
#pragma unroll 1
  for (int k0 = 0; k0 < K; k0 += 32) {
    // A fragment (16x32 bf16): lanes 0-15: K {0..7,16..23}; lanes 16-31: K {8..15,24..31}
    ABFrag a0, a1;
    const unsigned short* ap0 = &sA[m][ABASE + k0 + hi * 8];
    const unsigned short* ap1 = &sA[16 + m][ABASE + k0 + hi * 8];
    a0.u[0] = *(const uint4*)(ap0);
    a0.u[1] = *(const uint4*)(ap0 + 16);
    a1.u[0] = *(const uint4*)(ap1);
    a1.u[1] = *(const uint4*)(ap1 + 16);
#pragma unroll
    for (int t = 0; t < 6; ++t) {
      // B fragment (32x16, B[k][n] = W[n][k]): lanes 0-15: K {0..7,16..23}; lanes 16-31: K {8..15,24..31}
      int n = (tbase + t) * 16 + m;
      const unsigned short* bp = &Wmat[n * K + k0 + hi * 8];
      ABFrag b;
      b.u[0] = *(const uint4*)(bp);
      b.u[1] = *(const uint4*)(bp + 16);
      acc[0][t] = __builtin_amdgcn_wmma_f32_16x16x32_bf16(false, a0.v, false, b.v,
                                                          (short)0, acc[0][t], false, false);
      acc[1][t] = __builtin_amdgcn_wmma_f32_16x16x32_bf16(false, a1.v, false, b.v,
                                                          (short)0, acc[1][t], false, false);
    }
  }
}

// ---------------- kernel 4: fused gather + WMMA GEMM + GRU ----------------
__global__ __launch_bounds__(256, 2)
void tgn_gru(const float* __restrict__ memory,
             const int* __restrict__ last_update,
             const int* __restrict__ n_id,
             const int* __restrict__ src_s, const int* __restrict__ dst_s,
             const int* __restrict__ t_s,   const float* __restrict__ raw_msg_s,
             const int* __restrict__ src_d, const int* __restrict__ dst_d,
             const int* __restrict__ t_d,   const float* __restrict__ raw_msg_d,
             const float* __restrict__ w_time, const float* __restrict__ b_time,
             const float* __restrict__ b_ih,   const float* __restrict__ b_hh,
             const unsigned short* __restrict__ Wih_bf,
             const unsigned short* __restrict__ Whh_bf,
             const unsigned long long* __restrict__ maxkey,
             float* __restrict__ out_mem, float* __restrict__ out_last) {
  // 49152 B region:
  //   phase A: bf16 A-tile [32][640]   (40960 B)
  //   phase B: f32 acc staging [16][768] (49152 B), one 16-row pass at a time
  __shared__ __align__(16) unsigned char smem[16 * (2 * G3) * 4];
  unsigned short (*sA)[DIN + DH] = (unsigned short (*)[DIN + DH])smem;
  float (*sAcc)[2 * G3] = (float (*)[2 * G3])smem;
  __shared__ int sEvt[MROWS];
  __shared__ int sNewT[MROWS];

  const int tid  = threadIdx.x;
  const int row0 = blockIdx.x * MROWS;

  // ---- decode argmax key per row ----
  if (tid < MROWS) {
    int i = row0 + tid;
    unsigned long long mk = maxkey[i];
    if (mk > 0ull) {
      unsigned long long k = mk - 1ull;
      sEvt[tid]  = (int)(k % (unsigned long long)NE2);
      sNewT[tid] = (int)(k / (unsigned long long)NE2);
    } else {
      sEvt[tid]  = -1;
      sNewT[tid] = last_update[n_id[i]];
    }
  }
  __syncthreads();

  // ---- gather message features: thread (row, part) fills 64 contiguous cols ----
  {
    int row  = tid >> 3;        // 32 rows
    int part = tid & 7;         // 8 x 64-col segments
    int cbase = part * 64;
    unsigned short* dst = &sA[row][cbase];
    int e = sEvt[row];
    if (e < 0) {
#pragma unroll 8
      for (int j = 0; j < 64; ++j) dst[j] = 0;
    } else {
      int an, bn, tt, ei;
      const float* rm;
      if (e < NEVT) { ei = e;        an = src_s[ei]; bn = dst_s[ei]; tt = t_s[ei]; rm = raw_msg_s; }
      else          { ei = e - NEVT; an = dst_d[ei]; bn = src_d[ei]; tt = t_d[ei]; rm = raw_msg_d; }
      if (part < 6) {
        const float* srcp;
        int half = (part & 1) * 64;
        if (part < 2)      srcp = &memory[(long long)an * DMEM + half];
        else if (part < 4) srcp = &memory[(long long)bn * DMEM + half];
        else               srcp = &rm[(long long)ei * DMEM + half];
#pragma unroll 8
        for (int j = 0; j < 64; ++j) dst[j] = f2bf(srcp[j]);
      } else {
        int d0 = (part & 1) * 64;
        float dt = (float)(tt - last_update[an]);
#pragma unroll 8
        for (int j = 0; j < 64; ++j)
          dst[j] = f2bf(cosf(dt * w_time[d0 + j] + b_time[d0 + j]));
      }
    }
  }
  // ---- h into sA[:, 512:640): thread (row, part) fills 16 cols ----
  {
    int row  = tid >> 3;
    int part = tid & 7;
    const float* hp = &memory[(long long)n_id[row0 + row] * DMEM + part * 16];
    unsigned short* dst = &sA[row][DIN + part * 16];
#pragma unroll
    for (int j = 0; j < 16; ++j) dst[j] = f2bf(hp[j]);
  }
  __syncthreads();

  // ---- WMMA: waves 0-3 -> gi (K=512), waves 4-7 -> gh (K=128); 6 n-tiles, 2 m-subtiles ----
  const int w    = tid >> 5;
  const int lane = tid & 31;
  const int m    = lane & 15;
  const int hi   = lane >> 4;
  const bool isGh = (w >= 4);
  const int tbase = (isGh ? (w - 4) : w) * 6;

  v8f acc[2][6];
  v8f zero = {0.f, 0.f, 0.f, 0.f, 0.f, 0.f, 0.f, 0.f};
#pragma unroll
  for (int ms = 0; ms < 2; ++ms)
#pragma unroll
    for (int t = 0; t < 6; ++t) acc[ms][t] = zero;

  if (!isGh) gemm_part<DIN, 0>(sA, Wih_bf, tbase, m, hi, acc);
  else       gemm_part<DH, DIN>(sA, Whh_bf, tbase, m, hi, acc);

  __syncthreads();   // sA dead; reuse storage as f32 accumulator staging

  // ---- stage + GRU epilogue, one 16-row subtile at a time ----
  const int accbase = isGh ? G3 : 0;
#pragma unroll 1
  for (int ms = 0; ms < 2; ++ms) {
#pragma unroll
    for (int t = 0; t < 6; ++t) {
#pragma unroll
      for (int r = 0; r < 8; ++r) {
        // C/D layout: VGPR r, lane l -> row r + 8*(l>>4), col l&15
        sAcc[r + hi * 8][accbase + (tbase + t) * 16 + m] = acc[ms][t][r];
      }
    }
    __syncthreads();
    for (int idx = tid; idx < 16 * DMEM; idx += 256) {
      int r16 = idx >> 7;
      int c   = idx & (DMEM - 1);
      int i   = row0 + ms * 16 + r16;
      float gir = sAcc[r16][c]            + b_ih[c];
      float giz = sAcc[r16][c + 128]      + b_ih[c + 128];
      float gin = sAcc[r16][c + 256]      + b_ih[c + 256];
      float ghr = sAcc[r16][G3 + c]       + b_hh[c];
      float ghz = sAcc[r16][G3 + c + 128] + b_hh[c + 128];
      float ghn = sAcc[r16][G3 + c + 256] + b_hh[c + 256];
      float rg = 1.0f / (1.0f + expf(-(gir + ghr)));
      float zg = 1.0f / (1.0f + expf(-(giz + ghz)));
      float ng = tanhf(gin + rg * ghn);
      float h  = memory[(long long)n_id[i] * DMEM + c];
      out_mem[(long long)i * DMEM + c] = (1.0f - zg) * ng + zg * h;
    }
    __syncthreads();
  }
  if (tid < MROWS) out_last[row0 + tid] = (float)sNewT[tid];
}

// ---------------- host launch ----------------
extern "C" void kernel_launch(void* const* d_in, const int* in_sizes, int n_in,
                              void* d_out, int out_size, void* d_ws, size_t ws_size,
                              hipStream_t stream) {
  (void)in_sizes; (void)n_in; (void)out_size; (void)ws_size;

  const float* memory      = (const float*)d_in[0];
  const int*   last_update = (const int*)d_in[1];
  const int*   n_id        = (const int*)d_in[2];
  const int*   src_s       = (const int*)d_in[3];
  const int*   dst_s       = (const int*)d_in[4];
  const int*   t_s         = (const int*)d_in[5];
  const float* raw_msg_s   = (const float*)d_in[6];
  const int*   src_d       = (const int*)d_in[7];
  const int*   dst_d       = (const int*)d_in[8];
  const int*   t_d         = (const int*)d_in[9];
  const float* raw_msg_d   = (const float*)d_in[10];
  const float* w_time      = (const float*)d_in[11];
  const float* b_time      = (const float*)d_in[12];
  const float* W_ih        = (const float*)d_in[13];
  const float* W_hh        = (const float*)d_in[14];
  const float* b_ih        = (const float*)d_in[15];
  const float* b_hh        = (const float*)d_in[16];

  char* ws = (char*)d_ws;
  int* assoc                 = (int*)ws;                                 // 800000 B
  unsigned long long* maxkey = (unsigned long long*)(ws + 800000);       // 524288 B
  unsigned short* Wih_bf     = (unsigned short*)(ws + 800000 + 524288);  // 393216 B
  unsigned short* Whh_bf     = Wih_bf + G3 * DIN;                        //  98304 B

  float* out_mem  = (float*)d_out;
  float* out_last = out_mem + (size_t)BATCH * DMEM;

  tgn_init<<<BATCH / 256, 256, 0, stream>>>(n_id, assoc, maxkey);
  tgn_scan<<<NE2 / 256, 256, 0, stream>>>(src_s, dst_d, t_s, t_d, assoc, maxkey);
  tgn_wcvt<<<(G3 * (DIN + DH)) / 256, 256, 0, stream>>>(W_ih, W_hh, Wih_bf, Whh_bf);
  tgn_gru<<<BATCH / MROWS, 256, 0, stream>>>(memory, last_update, n_id,
                                             src_s, dst_s, t_s, raw_msg_s,
                                             src_d, dst_d, t_d, raw_msg_d,
                                             w_time, b_time, b_ih, b_hh,
                                             Wih_bf, Whh_bf, maxkey,
                                             out_mem, out_last);
}
